// VectorQuantizer_51694226375374
// MI455X (gfx1250) — matrix operand order repaired and checked by reading, not compiled, not run on hardware
//
#include <hip/hip_runtime.h>
#include <hip/hip_bf16.h>

// VQ argmin + gather + beta-blend for MI455X (gfx1250, wave32, WMMA bf16).
// score[n][k] = x_n . e_k - 0.5*|e_k|^2  (argmax == argmin of L2 distance)
// Per ktile, the full B tile (16 x global_load_b128) is issued as one clause
// so load latency retires progressively under the v_wmma bursts.

typedef __bf16 bf16_t;
typedef __attribute__((ext_vector_type(16))) __bf16 v16bf;
typedef __attribute__((ext_vector_type(8)))  __bf16 v8bf;
typedef __attribute__((ext_vector_type(8)))  float  v8f;

#define DDIM   256
#define KCB    2048
#define HWSZ   1024
#define NBATCH 32
#define NTOT   (NBATCH * HWSZ)      // 32768 query rows
#define ROWS   64                   // query rows per block (4 WMMA M-tiles)
#define PITCH  264                  // LDS row pitch in bf16 (528B: 16B aligned, conflict-free)
#define WAVES  8
#define KT_PER_WAVE (KCB / 16 / WAVES)  // 16 column tiles of 16 codes per wave

// ---------------- Kernel 1: codebook prep (fp32 -> bf16, bias = 0.5*|e|^2) --------
__global__ __launch_bounds__(256) void vq_prep(const float* __restrict__ E,
                                               bf16_t* __restrict__ Ebf,
                                               float* __restrict__ bias) {
  __shared__ float sums[8];
  const int k = blockIdx.x;
  const int t = threadIdx.x;
  float w = E[(size_t)k * DDIM + t];
  Ebf[(size_t)k * DDIM + t] = (bf16_t)w;
  float p = w * w;
#pragma unroll
  for (int off = 16; off >= 1; off >>= 1) p += __shfl_xor(p, off, 32);
  if ((t & 31) == 0) sums[t >> 5] = p;
  __syncthreads();
  if (t == 0) {
    float s = 0.f;
#pragma unroll
    for (int i = 0; i < 8; ++i) s += sums[i];
    bias[k] = 0.5f * s;
  }
}

// ---------------- Kernel 2: fused transpose + WMMA score + argmax + gather + blend -
__global__ __launch_bounds__(256) void vq_main(const float* __restrict__ lat,
                                               const bf16_t* __restrict__ Ebf,
                                               const float* __restrict__ bias,
                                               float* __restrict__ out,
                                               int* __restrict__ indOut) {
  __shared__ bf16_t Xlds[ROWS * PITCH];      // A staging, later reused for gathered q
  __shared__ float  redS[WAVES][4][16];
  __shared__ int    redI[WAVES][4][16];
  __shared__ int    indLds[ROWS];

  const int t   = threadIdx.x;
  const int n0  = blockIdx.x * ROWS;         // 64 consecutive n -> same batch image
  const int b   = n0 >> 10;
  const int hw0 = n0 & 1023;
  const float* latb = lat + (size_t)b * DDIM * HWSZ + hw0;

  // ---- Stage X rows: NCHW (d-major) -> LDS [row][d] bf16. Coalesced 256B reads.
  {
    const int r = t & 63;
    const int dbase = t >> 6;                // 0..3
#pragma unroll 4
    for (int i = 0; i < 64; ++i) {
      const int d = i * 4 + dbase;
      Xlds[r * PITCH + d] = (bf16_t)latb[d * HWSZ + r];
    }
  }
  __syncthreads();

  const int lane = t & 31;
  const int w    = t >> 5;
  const int ln   = lane & 15;                // C-layout column / A-layout row
  const int hi   = lane >> 4;                // lane half selector
  const int kt0  = w * KT_PER_WAVE;

  float bestS[4][8];
  int   bestI[4][8];
#pragma unroll
  for (int mt = 0; mt < 4; ++mt)
#pragma unroll
    for (int v = 0; v < 8; ++v) { bestS[mt][v] = -3.4e38f; bestI[mt][v] = 0; }

  // ---- Score loop: each wave owns 16 ktiles; B fragment reused across 4 M-tiles.
  for (int i = 0; i < KT_PER_WAVE; ++i) {
    const int code = (kt0 + i) * 16 + ln;
    const float bi = bias[code];
    const bf16_t* brow = Ebf + (size_t)code * DDIM + hi * 16;

    // Whole B tile for this ktile: lane holds K = hi*16..hi*16+15 per chunk,
    // 8 chunks => 16 independent global_load_b128 in one clause.
    v16bf btile[8];
#pragma unroll
    for (int c = 0; c < 8; ++c)
      btile[c] = *(const v16bf*)(brow + c * 32);

    v8f acc[4];
#pragma unroll
    for (int mt = 0; mt < 4; ++mt) acc[mt] = (v8f){0.f,0.f,0.f,0.f,0.f,0.f,0.f,0.f};

#pragma unroll
    for (int c = 0; c < 8; ++c) {            // D = 256 = 8 chunks of K=32
#pragma unroll
      for (int mt = 0; mt < 4; ++mt) {
        // A fragment: row ln of tile mt; halfs 0..15 = K hi*16..hi*16+15
        // (loop-invariant: compiler hoists all 32 fragments into VGPRs)
        const bf16_t* ap = &Xlds[(mt * 16 + ln) * PITCH + c * 32 + hi * 16];
        const v8bf lo  = *(const v8bf*)ap;
        const v8bf hi8 = *(const v8bf*)(ap + 8);
        const v16bf afr = __builtin_shufflevector(
            lo, hi8, 0, 1, 2, 3, 4, 5, 6, 7, 8, 9, 10, 11, 12, 13, 14, 15);
        acc[mt] = __builtin_amdgcn_wmma_f32_16x16x32_bf16(
            false, afr, false, btile[c], (short)0, acc[mt], false, false);
      }
    }
#pragma unroll
    for (int mt = 0; mt < 4; ++mt)
#pragma unroll
      for (int v = 0; v < 8; ++v) {
        const float s = acc[mt][v] - bi;
        if (s > bestS[mt][v] || (s == bestS[mt][v] && code < bestI[mt][v])) {
          bestS[mt][v] = s; bestI[mt][v] = code;
        }
      }
  }

  // ---- Lane butterfly within each 16-lane half (rows M = v + 8*hi)
#pragma unroll
  for (int off = 8; off >= 1; off >>= 1) {
#pragma unroll
    for (int mt = 0; mt < 4; ++mt)
#pragma unroll
      for (int v = 0; v < 8; ++v) {
        const float os = __shfl_xor(bestS[mt][v], off, 32);
        const int   oi = __shfl_xor(bestI[mt][v], off, 32);
        if (os > bestS[mt][v] || (os == bestS[mt][v] && oi < bestI[mt][v])) {
          bestS[mt][v] = os; bestI[mt][v] = oi;
        }
      }
  }
  if (ln == 0) {
#pragma unroll
    for (int mt = 0; mt < 4; ++mt)
#pragma unroll
      for (int v = 0; v < 8; ++v) {
        redS[w][mt][hi * 8 + v] = bestS[mt][v];
        redI[w][mt][hi * 8 + v] = bestI[mt][v];
      }
  }
  __syncthreads();

  // ---- Cross-wave reduce: one thread per row
  if (t < ROWS) {
    const int mt = t >> 4, m = t & 15;
    float s = redS[0][mt][m];
    int idx = redI[0][mt][m];
#pragma unroll
    for (int ww = 1; ww < WAVES; ++ww) {
      const float os = redS[ww][mt][m];
      const int   oi = redI[ww][mt][m];
      if (os > s || (os == s && oi < idx)) { s = os; idx = oi; }
    }
    indLds[t] = idx;
    indOut[n0 + t] = idx;
  }
  __syncthreads();

  // ---- Gather winning codebook rows (bf16) into Xlds (reuse), coalesced 16B chunks
#pragma unroll
  for (int j = 0; j < 8; ++j) {
    const int id = j * 256 + t;
    const int r  = id >> 5;                  // 32 chunks of 16B per 512B row
    const int cj = id & 31;
    const int idx = indLds[r];
    *(v8bf*)(Xlds + r * PITCH + cj * 8) =
        *(const v8bf*)(Ebf + (size_t)idx * DDIM + cj * 8);
  }
  __syncthreads();

  // ---- Blend + NCHW write: out = x + 0.25*(q - x); hw-fast => coalesced stores
  float* outb = out + (size_t)b * DDIM * HWSZ + hw0;
  {
    const int r = t & 63;
    const int dbase = t >> 6;
#pragma unroll 4
    for (int i = 0; i < 64; ++i) {
      const int d = i * 4 + dbase;
      const float x = latb[d * HWSZ + r];
      const float q = (float)Xlds[r * PITCH + d];
      outb[d * HWSZ + r] = x + 0.25f * (q - x);
    }
  }
}

extern "C" void kernel_launch(void* const* d_in, const int* in_sizes, int n_in,
                              void* d_out, int out_size, void* d_ws, size_t ws_size,
                              hipStream_t stream) {
  const float* lat = (const float*)d_in[0];   // (32,256,32,32) fp32
  const float* E   = (const float*)d_in[1];   // (2048,256) fp32

  bf16_t* Ebf  = (bf16_t*)d_ws;                                    // 1 MB
  float*  bias = (float*)((char*)d_ws + (size_t)KCB * DDIM * 2);   // 8 KB

  float* out = (float*)d_out;                                      // 8,388,608 f32
  int*   ind = (int*)d_out + (size_t)NBATCH * DDIM * HWSZ;         // 32,768 i32

  vq_prep<<<KCB, 256, 0, stream>>>(E, Ebf, bias);
  vq_main<<<NTOT / ROWS, 256, 0, stream>>>(lat, Ebf, bias, out, ind);
}
